// ELA_26302379721416
// MI455X (gfx1250) — matrix operand order repaired and checked by reading, not compiled, not run on hardware
//
#include <hip/hip_runtime.h>
#include <hip/hip_bf16.h>

// ---------------------------------------------------------------------------
// ELA forward for MI455X (gfx1250, wave32, WMMA).
// GEMMs run on v_wmma_f32_16x16x32_bf16 (fp32 accumulate).
// Weights are transposed+converted to bf16 once so the recurrence working set
// (Rt1+Rt2 = 151 MB bf16) fits the 192 MB L2.
// ---------------------------------------------------------------------------

typedef __bf16 bf16;
typedef __attribute__((ext_vector_type(16))) __bf16 v16bf;
typedef __attribute__((ext_vector_type(8)))  __bf16 v8bf;
typedef __attribute__((ext_vector_type(8)))  float  v8f;

#define Dd    3072      // feature dim (C*W)
#define ND    12288     // 4*D
#define BB    16        // batch
#define TT    48        // sequence length (H == W)
#define CC    64        // channels
#define HW    48        // spatial
#define ROWS  768       // B*T
#define NELEM_X (BB*CC*HW*HW)          // 2359296 == ROWS*Dd
#define WELEM ((size_t)Dd * (size_t)ND)

// ---- WMMA operand loaders --------------------------------------------------
// A tile 16x32 (MxK), bf16 row-major, per ISA layout:
//   lanes 0-15 : M=lane,    K = {k0..k0+7, k0+16..k0+23}
//   lanes 16-31: M=lane-16, K = {k0+8..k0+15, k0+24..k0+31}
__device__ __forceinline__ v16bf load_a16x32(const bf16* __restrict__ A, int lda,
                                             int m0, int k0, int lane)
{
  const int half = lane >> 4, r = lane & 15;
  const bf16* p = A + (size_t)(m0 + r) * lda + k0 + half * 8;
  v8bf lo = *(const v8bf*)p;
  v8bf hi = *(const v8bf*)(p + 16);
  v16bf a;
#pragma unroll
  for (int i = 0; i < 8; ++i) { a[i] = lo[i]; a[8 + i] = hi[i]; }
  return a;
}

// B tile 32x16 (KxN) loaded from a PRE-TRANSPOSED [N][K] bf16 matrix (two 16-byte loads):
// lane's K-values match the A layout:
//   lanes 0-15 : N=n0+lane, K = {k0..k0+7, k0+16..k0+23}
//   lanes 16-31: N=n0+lane-16, K = {k0+8..k0+15, k0+24..k0+31}
__device__ __forceinline__ v16bf load_bT32x16(const bf16* __restrict__ Bt, int ldb,
                                              int n0, int k0, int lane)
{
  const int kh = lane >> 4, n = lane & 15;
  const bf16* p = Bt + (size_t)(n0 + n) * ldb + k0 + kh * 8;
  v8bf lo = *(const v8bf*)p;
  v8bf hi = *(const v8bf*)(p + 16);
  v16bf b;
#pragma unroll
  for (int i = 0; i < 8; ++i) { b[i] = lo[i]; b[8 + i] = hi[i]; }
  return b;
}

// ---- 1) transpose + fp32->bf16 conversion of W1,R1,W2,R2 -------------------
// src: [D][4D] fp32 row-major  ->  dst: [4D][D] bf16 row-major
__global__ __launch_bounds__(256) void k_tcvt(
    const float* __restrict__ W1, const float* __restrict__ R1,
    const float* __restrict__ W2, const float* __restrict__ R2,
    bf16* __restrict__ Wt1, bf16* __restrict__ Rt1,
    bf16* __restrict__ Wt2, bf16* __restrict__ Rt2)
{
  __shared__ float tile[32][33];
  const int mi = blockIdx.z;
  const float* src = (mi == 0) ? W1 : (mi == 1) ? R1 : (mi == 2) ? W2 : R2;
  bf16*       dst = (mi == 0) ? Wt1 : (mi == 1) ? Rt1 : (mi == 2) ? Wt2 : Rt2;
  const int x0 = blockIdx.x * 32;   // n in [0,12288)
  const int y0 = blockIdx.y * 32;   // d in [0,3072)
  const int tx = threadIdx.x, ty = threadIdx.y;
#pragma unroll
  for (int j = 0; j < 4; ++j)
    tile[ty + j * 8][tx] = src[(size_t)(y0 + ty + j * 8) * ND + x0 + tx];
  __syncthreads();
#pragma unroll
  for (int j = 0; j < 4; ++j)
    dst[(size_t)(x0 + ty + j * 8) * Dd + y0 + tx] = (bf16)tile[tx][ty + j * 8];
}

// ---- 2) x fp32 -> bf16 ------------------------------------------------------
__global__ __launch_bounds__(256) void k_cvtx(const float* __restrict__ x,
                                              bf16* __restrict__ xb, int n)
{
  int i = blockIdx.x * blockDim.x + threadIdx.x;
  if (i < n) xb[i] = (bf16)x[i];
}

// ---- 3) pre = X @ W + b  (768x3072 @ 3072x12288, both stacks) ---------------
__global__ __launch_bounds__(32) void k_pregemm(
    const bf16* __restrict__ xb,
    const bf16* __restrict__ Wt1, const bf16* __restrict__ Wt2,
    const float* __restrict__ b1, const float* __restrict__ b2,
    float* __restrict__ pre1, float* __restrict__ pre2)
{
  const int lane = threadIdx.x;
  const int n0 = blockIdx.x * 16;
  const int m0 = blockIdx.y * 16;
  const int s  = blockIdx.z;
  const bf16*  Wt   = s ? Wt2 : Wt1;
  const float* bias = s ? b2  : b1;
  float*       pre  = s ? pre2 : pre1;

  v8f acc = {};
  for (int k0 = 0; k0 < Dd; k0 += 32) {
    v16bf a = load_a16x32(xb, Dd, m0, k0, lane);
    v16bf b = load_bT32x16(Wt, Dd, n0, k0, lane);
    if (k0 + 32 < Dd)  // hint next B tile (global_prefetch_b8)
      __builtin_prefetch(Wt + (size_t)(n0 + (lane & 15)) * Dd + k0 + 32, 0, 3);
    acc = __builtin_amdgcn_wmma_f32_16x16x32_bf16(false, a, false, b,
                                                  (short)0, acc, false, false);
  }
  const int col = lane & 15;
  const int rb  = (lane >> 4) * 8;    // C/D layout: VGPR i -> row rb+i
  const float bv = bias[n0 + col];
#pragma unroll
  for (int i = 0; i < 8; ++i)
    pre[(size_t)(m0 + rb + i) * ND + n0 + col] = acc[i] + bv;
}

// ---- 4) one sLSTM time step (both stacks): g = pre[:,t,:] + h@R, gate -------
// grid (192, 2); block 128 (4 waves). Wave w reduces K range [w*768, w*768+768)
// for its 16-column feature tile across the 4 gates; partials reduced via LDS.
__global__ __launch_bounds__(128) void k_step(
    const bf16* __restrict__ Rt1, const bf16* __restrict__ Rt2,
    const float* __restrict__ pre1, const float* __restrict__ pre2,
    float* __restrict__ y1, float* __restrict__ y2,
    bf16* __restrict__ hbuf,   // [2 stacks][2 parity bufs][16][3072] bf16
    float* __restrict__ st,    // [2 stacks][3 (c,n,m)][16][3072] fp32
    int t)
{
  const int tid = threadIdx.x, lane = tid & 31, wv = tid >> 5;
  const int d0 = blockIdx.x * 16;
  const int s  = blockIdx.y;
  const bf16*  Rt  = s ? Rt2  : Rt1;
  const float* pre = s ? pre2 : pre1;
  float*       y   = s ? y2   : y1;
  bf16* hrd = hbuf + ((size_t)s * 2 + ((t + 1) & 1)) * (size_t)(BB * Dd);
  bf16* hwr = hbuf + ((size_t)s * 2 + (t & 1))       * (size_t)(BB * Dd);
  float* stc = st + (size_t)s * 3 * BB * Dd;

  __shared__ float red[4][4][256];   // [wave][gate][16x16 tile]

  v8f acc[4] = {};
  if (t > 0) {
    const int kb = wv * (Dd / 4);
    for (int kk = 0; kk < Dd / 4; kk += 32) {
      v16bf a = load_a16x32(hrd, Dd, 0, kb + kk, lane);
#pragma unroll
      for (int g = 0; g < 4; ++g) {
        v16bf b = load_bT32x16(Rt, Dd, g * Dd + d0, kb + kk, lane);
        if (kk + 32 < Dd / 4)
          __builtin_prefetch(Rt + (size_t)(g * Dd + d0 + (lane & 15)) * Dd + kb + kk + 32, 0, 3);
        acc[g] = __builtin_amdgcn_wmma_f32_16x16x32_bf16(false, a, false, b,
                                                         (short)0, acc[g], false, false);
      }
    }
  }
  const int col = lane & 15, rb = (lane >> 4) * 8;
#pragma unroll
  for (int g = 0; g < 4; ++g)
#pragma unroll
    for (int i = 0; i < 8; ++i)
      red[wv][g][(rb + i) * 16 + col] = acc[g][i];
  __syncthreads();

  for (int e = tid; e < 256; e += 128) {
    const int mm = e >> 4;          // batch row 0..15
    const int d  = d0 + (e & 15);   // feature
    const size_t prow = ((size_t)mm * TT + t) * ND;
    float it = pre[prow + 0 * Dd + d];
    float ft = pre[prow + 1 * Dd + d];
    float zt = pre[prow + 2 * Dd + d];
    float ot = pre[prow + 3 * Dd + d];
#pragma unroll
    for (int w = 0; w < 4; ++w) {
      it += red[w][0][e]; ft += red[w][1][e];
      zt += red[w][2][e]; ot += red[w][3][e];
    }
    const size_t si = (size_t)mm * Dd + d;
    float c = 0.f, n = 0.f, m = 0.f;
    if (t > 0) { c = stc[si]; n = stc[(size_t)BB * Dd + si]; m = stc[2 * (size_t)BB * Dd + si]; }
    const float m_new = fmaxf(ft + m, it);          // stabilizer
    const float ie = __expf(it - m_new);
    const float fe = __expf(ft + m - m_new);
    const float c_new = fe * c + ie * tanhf(zt);
    const float n_new = fe * n + ie;
    const float sig_o = 1.f / (1.f + __expf(-ot));
    const float h_new = sig_o * c_new / (n_new + 1e-6f);
    stc[si] = c_new;
    stc[(size_t)BB * Dd + si] = n_new;
    stc[2 * (size_t)BB * Dd + si] = m_new;
    hwr[si] = (bf16)h_new;                          // next-step A operand
    y[((size_t)mm * TT + t) * Dd + d] = h_new;      // fp32 output stream
  }
}

// ---- 5) directional mean pools ---------------------------------------------
// xh[b][c][p] = mean_h y2_flat[b][c*2304 + p*48 + h]   (contiguous 48)
// xw[b][c][p] = mean_w y1_flat[b][c*2304 + w*48 + p]   (stride 48)
__global__ __launch_bounds__(256) void k_pool(
    const float* __restrict__ y1, const float* __restrict__ y2,
    float* __restrict__ xh, float* __restrict__ xw)
{
  const int NP = BB * CC * HW;
  int idx = blockIdx.x * blockDim.x + threadIdx.x;
  if (idx >= 2 * NP) return;
  const int which = idx / NP;
  const int r = idx % NP;
  const int b = r / (CC * HW);
  const int c = (r / HW) % CC;
  const int p = r % HW;
  float sum = 0.f;
  if (which == 0) {
    const float* base = y2 + (size_t)b * (CC * HW * HW) + c * (HW * HW) + p * HW;
    for (int q = 0; q < HW; ++q) sum += base[q];
    xh[r] = sum * (1.f / HW);
  } else {
    const float* base = y1 + (size_t)b * (CC * HW * HW) + c * (HW * HW) + p;
    for (int q = 0; q < HW; ++q) sum += base[q * HW];
    xw[r] = sum * (1.f / HW);
  }
}

// ---- 6) grouped conv1d(K=5,G=8) + GroupNorm(16) + sigmoid -------------------
// One block per (batch, signal). Whole [64][48] signal lives in LDS.
__global__ __launch_bounds__(256) void k_convgn(
    const float* __restrict__ xh, const float* __restrict__ xw,
    const float* __restrict__ cw, const float* __restrict__ gamma,
    const float* __restrict__ beta,
    float* __restrict__ sigh, float* __restrict__ sigw)
{
  const int b = blockIdx.x;
  const int which = blockIdx.y;
  const float* in  = (which == 0 ? xh : xw)   + (size_t)b * CC * HW;
  float*       out = (which == 0 ? sigh : sigw) + (size_t)b * CC * HW;
  __shared__ float sin_[CC][HW];
  __shared__ float scv[CC][HW];
  __shared__ float smu[16], srs[16];
  const int tid = threadIdx.x;
  for (int i = tid; i < CC * HW; i += 256) sin_[i / HW][i % HW] = in[i];
  __syncthreads();
  for (int i = tid; i < CC * HW; i += 256) {
    const int o = i / HW, l = i % HW;
    const int g = o >> 3;                    // conv group (8 in-channels)
    float acc = 0.f;
#pragma unroll
    for (int ci = 0; ci < 8; ++ci) {
      const float* wrow = cw + ((size_t)o * 8 + ci) * 5;
#pragma unroll
      for (int k = 0; k < 5; ++k) {
        const int ll = l + k - 2;
        if (ll >= 0 && ll < HW) acc += sin_[g * 8 + ci][ll] * wrow[k];
      }
    }
    scv[o][l] = acc;
  }
  __syncthreads();
  if (tid < 16) {                            // GroupNorm stats: 4 ch x 48
    float s = 0.f, ss = 0.f;
    const int c0 = tid * 4;
    for (int c = c0; c < c0 + 4; ++c)
      for (int l = 0; l < HW; ++l) { float v = scv[c][l]; s += v; ss += v * v; }
    const float mu = s * (1.f / 192.f);
    smu[tid] = mu;
    srs[tid] = rsqrtf(ss * (1.f / 192.f) - mu * mu + 1e-5f);
  }
  __syncthreads();
  for (int i = tid; i < CC * HW; i += 256) {
    const int c = i / HW;
    const int grp = c >> 2;
    float v = (scv[c][i % HW] - smu[grp]) * srs[grp];
    v = v * gamma[c] + beta[c];
    out[i] = 1.f / (1.f + __expf(-v));
  }
}

// ---- 7) out = sig_h[b,c,h] * sig_w[b,c,w] * x -------------------------------
__global__ __launch_bounds__(256) void k_final(
    const float* __restrict__ x, const float* __restrict__ sigh,
    const float* __restrict__ sigw, float* __restrict__ out)
{
  int idx = blockIdx.x * blockDim.x + threadIdx.x;
  if (idx >= NELEM_X) return;
  const int w = idx % HW;
  const int h = (idx / HW) % HW;
  const int c = (idx / (HW * HW)) % CC;
  const int b = idx / (CC * HW * HW);
  out[idx] = sigh[((size_t)b * CC + c) * HW + h] *
             sigw[((size_t)b * CC + c) * HW + w] * x[idx];
}

// ---------------------------------------------------------------------------
extern "C" void kernel_launch(void* const* d_in, const int* in_sizes, int n_in,
                              void* d_out, int out_size, void* d_ws, size_t ws_size,
                              hipStream_t stream)
{
  (void)in_sizes; (void)n_in; (void)out_size; (void)ws_size;
  const float* x     = (const float*)d_in[0];
  const float* W1    = (const float*)d_in[1];
  const float* R1    = (const float*)d_in[2];
  const float* b1    = (const float*)d_in[3];
  const float* W2    = (const float*)d_in[4];
  const float* R2    = (const float*)d_in[5];
  const float* b2    = (const float*)d_in[6];
  const float* convw = (const float*)d_in[7];
  const float* gam   = (const float*)d_in[8];
  const float* bet   = (const float*)d_in[9];
  float* out = (float*)d_out;
  char* ws = (char*)d_ws;

  size_t off = 0;
  auto take = [&off, ws](size_t bytes) -> char* {
    char* p = ws + off;
    off += (bytes + 255) & ~(size_t)255;
    return p;
  };
  bf16* Wt1 = (bf16*)take(WELEM * sizeof(bf16));
  bf16* Rt1 = (bf16*)take(WELEM * sizeof(bf16));
  bf16* Wt2 = (bf16*)take(WELEM * sizeof(bf16));
  bf16* Rt2 = (bf16*)take(WELEM * sizeof(bf16));
  bf16* xb  = (bf16*)take((size_t)ROWS * Dd * sizeof(bf16));
  float* pre1 = (float*)take((size_t)ROWS * ND * sizeof(float));
  float* pre2 = (float*)take((size_t)ROWS * ND * sizeof(float));
  float* y1   = (float*)take((size_t)ROWS * Dd * sizeof(float));
  float* y2   = (float*)take((size_t)ROWS * Dd * sizeof(float));
  bf16*  hbuf = (bf16*)take((size_t)2 * 2 * BB * Dd * sizeof(bf16));
  float* st   = (float*)take((size_t)2 * 3 * BB * Dd * sizeof(float));
  float* xh   = (float*)take((size_t)BB * CC * HW * sizeof(float));
  float* xw   = (float*)take((size_t)BB * CC * HW * sizeof(float));
  float* sigh = (float*)take((size_t)BB * CC * HW * sizeof(float));
  float* sigw = (float*)take((size_t)BB * CC * HW * sizeof(float));

  // 1) transpose + convert weights to bf16 (makes both R's L2-resident)
  k_tcvt<<<dim3(ND / 32, Dd / 32, 4), dim3(32, 8), 0, stream>>>(
      W1, R1, W2, R2, Wt1, Rt1, Wt2, Rt2);
  // 2) activations to bf16
  k_cvtx<<<(NELEM_X + 255) / 256, 256, 0, stream>>>(x, xb, NELEM_X);
  // 3) input projections for all timesteps, both stacks (WMMA)
  k_pregemm<<<dim3(ND / 16, ROWS / 16, 2), 32, 0, stream>>>(
      xb, Wt1, Wt2, b1, b2, pre1, pre2);
  // 4) 48 sequential recurrence steps (WMMA h@R + exp gating)
  for (int t = 0; t < TT; ++t)
    k_step<<<dim3(Dd / 16, 2), 128, 0, stream>>>(
        Rt1, Rt2, pre1, pre2, y1, y2, hbuf, st, t);
  // 5) directional pooling
  k_pool<<<(2 * BB * CC * HW + 255) / 256, 256, 0, stream>>>(y1, y2, xh, xw);
  // 6) conv1d(g=8,k=5) + GroupNorm(16) + sigmoid
  k_convgn<<<dim3(BB, 2), 256, 0, stream>>>(xh, xw, convw, gam, bet, sigh, sigw);
  // 7) final modulation
  k_final<<<(NELEM_X + 255) / 256, 256, 0, stream>>>(x, sigh, sigw, out);
}
